// GlobalFeatureEncoder_53695681135024
// MI455X (gfx1250) — matrix operand order repaired and checked by reading, not compiled, not run on hardware
//
#include <hip/hip_runtime.h>
#include <hip/hip_bf16.h>
#include <math.h>

typedef __attribute__((ext_vector_type(16))) _Float16 v16h;
typedef __attribute__((ext_vector_type(8)))  _Float16 v8h;
typedef __attribute__((ext_vector_type(8)))  float    v8f;

#define HID 256
#define NB  8192
#define G4  1024          // 4*HID
#define M_TILE 32         // batch rows per workgroup (2 WMMA m-tiles)

struct RunDesc {
    const float* wih;     // [1024] (input size is 1)
    const float* bih;     // [1024]
    const float* bhh;     // [1024]
    int seq_base;         // column offset in x row
    int seq_len;
    int reversed;
};

struct LstmArgs {
    RunDesc runs[4];
    const float*    x;       // [8192][28]
    const _Float16* whh16;   // [4][1024][256] f16
    float*          hsum;    // [8192][256] accumulator
};

struct PrepArgs {
    const float* whh[4];
    _Float16*    whh16;
    float*       hsum;
};

__device__ __forceinline__ float sigm(float v) { return 1.0f / (1.0f + __expf(-v)); }

// ---------------------------------------------------------------------------
// prep: fp32 -> f16 weight conversion + zero the accumulation buffer
// ---------------------------------------------------------------------------
__global__ __launch_bounds__(256) void prep_kernel(PrepArgs a) {
    size_t idx = (size_t)blockIdx.x * 256 + threadIdx.x;
    const size_t wtot = (size_t)4 * G4 * HID;        // 1,048,576
    if (idx < wtot) {
        int run = (int)(idx / (G4 * HID));
        int off = (int)(idx % (G4 * HID));
        a.whh16[idx] = (_Float16)a.whh[run][off];
    }
    if (idx < (size_t)NB * HID) {                    // 2,097,152
        a.hsum[idx] = 0.0f;
    }
}

// ---------------------------------------------------------------------------
// lstm: one workgroup = 32 batch rows x one (lstm, direction) run.
// 8 waves; wave w owns gate tiles {jt, jt+16, jt+32, jt+48} for jt in {w, w+8}
// so i/f/g/o for each (m, j) pair live in one lane -> pointwise update in
// registers with per-lane persistent c state. Each B fragment (whh rows from
// L2) is reused for the two batch m-tiles, halving L2 weight traffic.
// ---------------------------------------------------------------------------
__global__ __launch_bounds__(256) void lstm_kernel(LstmArgs args) {
    const int run    = blockIdx.y;
    const RunDesc rd = args.runs[run];
    const int batch0 = blockIdx.x * M_TILE;
    const int tid  = threadIdx.x;
    const int lane = tid & 31;
    const int wave = tid >> 5;
    const int hs   = lane >> 4;     // lane-half select
    const int nl   = lane & 15;     // N / M position within tile

    __shared__ _Float16 ldsH[M_TILE * HID];   // h state, f16, 16 KB
    __shared__ float    ldsX[16 * M_TILE];    // per-step sequence scalars, 2 KB

    // preload sequence scalars (handles reversal)
    for (int i = tid; i < rd.seq_len * M_TILE; i += 256) {
        int t = i / M_TILE, m = i % M_TILE;
        int tt = rd.reversed ? (rd.seq_len - 1 - t) : t;
        ldsX[t * M_TILE + m] = args.x[(size_t)(batch0 + m) * 28 + rd.seq_base + tt];
    }
    // h0 = 0
    for (int i = tid; i < M_TILE * HID; i += 256) ldsH[i] = (_Float16)0.0f;

    // loop-invariant per-lane data: weight-row pointers, wih, fused bias
    const _Float16* wbase = args.whh16 + (size_t)run * G4 * HID;
    const _Float16* brow[2][4];
    float wih_r[2][4], bias_r[2][4];
#pragma unroll
    for (int ji = 0; ji < 2; ++ji) {
        int j = (wave + 8 * ji) * 16 + nl;           // hidden index 0..255
#pragma unroll
        for (int g = 0; g < 4; ++g) {
            int n = j + g * HID;                     // gate row 0..1023
            brow[ji][g]   = wbase + (size_t)n * HID; // whh row n (contiguous K)
            wih_r[ji][g]  = rd.wih[n];
            bias_r[ji][g] = rd.bih[n] + rd.bhh[n];
        }
    }

    float c_st[2][2][8];                             // [mt][ji][r]
#pragma unroll
    for (int mt = 0; mt < 2; ++mt)
#pragma unroll
        for (int ji = 0; ji < 2; ++ji)
#pragma unroll
            for (int r = 0; r < 8; ++r) c_st[mt][ji][r] = 0.0f;

    __syncthreads();

    for (int t = 0; t < rd.seq_len; ++t) {
        // ---- gates = h @ whh.T via WMMA (M=32 batch, N=1024 gates, K=256) ----
        v8f acc[2][2][4];                            // [mt][ji][gate]
#pragma unroll
        for (int mt = 0; mt < 2; ++mt)
#pragma unroll
            for (int ji = 0; ji < 2; ++ji)
#pragma unroll
                for (int g = 0; g < 4; ++g) acc[mt][ji][g] = (v8f){0,0,0,0,0,0,0,0};

#pragma unroll
        for (int kc = 0; kc < 8; ++kc) {            // K chunks of 32
            // A fragments (16x32 f16): lane M=nl; K = hs*16+e
            v16h a[2];
#pragma unroll
            for (int mt = 0; mt < 2; ++mt) {
                const _Float16* ap = &ldsH[(mt * 16 + nl) * HID + kc * 32 + hs * 16];
                v8h a0 = *(const v8h*)ap;
                v8h a1 = *(const v8h*)(ap + 8);
#pragma unroll
                for (int e = 0; e < 8; ++e) { a[mt][e] = a0[e]; a[mt][e + 8] = a1[e]; }
            }

#pragma unroll
            for (int ji = 0; ji < 2; ++ji)
#pragma unroll
                for (int g = 0; g < 4; ++g) {
                    // B fragment (32x16 f16): lane column N=nl,
                    // 16 contiguous K values per lane (K = hs*16 + e)
                    const _Float16* bp = brow[ji][g] + kc * 32 + hs * 16;
                    v16h b = *(const v16h*)bp;
#pragma unroll
                    for (int mt = 0; mt < 2; ++mt)   // reuse B for both m-tiles
                        acc[mt][ji][g] = __builtin_amdgcn_wmma_f32_16x16x32_f16(
                            false, a[mt], false, b, (short)0, acc[mt][ji][g], false, false);
                }
        }
        __syncthreads();   // all waves done reading old h from LDS

        // ---- pointwise LSTM update, entirely in registers ----
#pragma unroll
        for (int mt = 0; mt < 2; ++mt)
#pragma unroll
            for (int ji = 0; ji < 2; ++ji) {
                int j = (wave + 8 * ji) * 16 + nl;
#pragma unroll
                for (int r = 0; r < 8; ++r) {
                    int m = mt * 16 + r + 8 * hs;    // C/D layout: M = r + 8*half
                    float xt = ldsX[t * M_TILE + m];
                    float gi = acc[mt][ji][0][r] + xt * wih_r[ji][0] + bias_r[ji][0];
                    float gf = acc[mt][ji][1][r] + xt * wih_r[ji][1] + bias_r[ji][1];
                    float gg = acc[mt][ji][2][r] + xt * wih_r[ji][2] + bias_r[ji][2];
                    float go = acc[mt][ji][3][r] + xt * wih_r[ji][3] + bias_r[ji][3];
                    float c  = sigm(gf) * c_st[mt][ji][r] + sigm(gi) * tanhf(gg);
                    float h  = sigm(go) * tanhf(c);
                    c_st[mt][ji][r] = c;
                    ldsH[m * HID + j] = (_Float16)h; // next step's A matrix
                }
            }
        __syncthreads();
    }

    // accumulate final h across the 4 runs (read back the f16 we just wrote;
    // each lane reads only addresses it wrote itself)
#pragma unroll
    for (int mt = 0; mt < 2; ++mt)
#pragma unroll
        for (int ji = 0; ji < 2; ++ji) {
            int j = (wave + 8 * ji) * 16 + nl;
#pragma unroll
            for (int r = 0; r < 8; ++r) {
                int m = mt * 16 + r + 8 * hs;
                float h = (float)ldsH[m * HID + j];
                atomicAdd(&args.hsum[(size_t)(batch0 + m) * HID + j], h);
            }
        }
}

// ---------------------------------------------------------------------------
// final: emb_s + hsum, LayerNorm over 256
// ---------------------------------------------------------------------------
__global__ __launch_bounds__(256) void final_kernel(
    const float* __restrict__ x, const float* __restrict__ w_s,
    const float* __restrict__ b_s, const float* __restrict__ gamma,
    const float* __restrict__ beta, const float* __restrict__ hsum,
    float* __restrict__ out) {
    const int b = blockIdx.x;
    const int j = threadIdx.x;
    __shared__ float red[256];
    __shared__ float xs[4];
    if (j < 4) xs[j] = x[(size_t)b * 28 + j];
    __syncthreads();

    float g = hsum[(size_t)b * HID + j] + b_s[j];
#pragma unroll
    for (int k = 0; k < 4; ++k) g += xs[k] * w_s[j * 4 + k];

    red[j] = g;
    __syncthreads();
    for (int s = 128; s > 0; s >>= 1) { if (j < s) red[j] += red[j + s]; __syncthreads(); }
    float mu = red[0] * (1.0f / HID);
    __syncthreads();
    float d = g - mu;
    red[j] = d * d;
    __syncthreads();
    for (int s = 128; s > 0; s >>= 1) { if (j < s) red[j] += red[j + s]; __syncthreads(); }
    float var = red[0] * (1.0f / HID);
    out[(size_t)b * HID + j] = d * rsqrtf(var + 1e-5f) * gamma[j] + beta[j];
}

// ---------------------------------------------------------------------------
extern "C" void kernel_launch(void* const* d_in, const int* in_sizes, int n_in,
                              void* d_out, int out_size, void* d_ws, size_t ws_size,
                              hipStream_t stream) {
    const float* x     = (const float*)d_in[0];
    const float* w_s   = (const float*)d_in[1];
    const float* b_s   = (const float*)d_in[2];
    const float* gamma = (const float*)d_in[19];
    const float* beta  = (const float*)d_in[20];

    // ws layout: [0, 2MB) whh f16 x4 ; [2MB, 10MB) fp32 h accumulator
    _Float16* whh16 = (_Float16*)d_ws;
    float*    hsum  = (float*)((char*)d_ws + (size_t)4 * G4 * HID * sizeof(_Float16));

    PrepArgs pa;
    pa.whh[0] = (const float*)d_in[4];   // wA_hh_f
    pa.whh[1] = (const float*)d_in[8];   // wA_hh_r
    pa.whh[2] = (const float*)d_in[12];  // wB_hh_f
    pa.whh[3] = (const float*)d_in[16];  // wB_hh_r
    pa.whh16 = whh16;
    pa.hsum  = hsum;
    prep_kernel<<<(NB * HID) / 256, 256, 0, stream>>>(pa);

    LstmArgs la;
    la.x = x; la.whh16 = whh16; la.hsum = hsum;
    // run 0: A forward, 1: A reverse, 2: B forward, 3: B reverse
    la.runs[0] = { (const float*)d_in[3],  (const float*)d_in[5],  (const float*)d_in[6],  4, 9, 0 };
    la.runs[1] = { (const float*)d_in[7],  (const float*)d_in[9],  (const float*)d_in[10], 4, 9, 1 };
    la.runs[2] = { (const float*)d_in[11], (const float*)d_in[13], (const float*)d_in[14], 13, 15, 0 };
    la.runs[3] = { (const float*)d_in[15], (const float*)d_in[17], (const float*)d_in[18], 13, 15, 1 };
    lstm_kernel<<<dim3(NB / M_TILE, 4), 256, 0, stream>>>(la);

    final_kernel<<<NB, 256, 0, stream>>>(x, w_s, b_s, gamma, beta, hsum, (float*)d_out);
}
